// GATClassifier_9801115369514
// MI455X (gfx1250) — matrix operand order repaired and checked by reading, not compiled, not run on hardware
//
#include <hip/hip_runtime.h>
#include <hip/hip_bf16.h>
#include <math.h>

// ---------------- problem constants (match reference) ----------------
#define NNODES   100000
#define NEDGES   1600000
#define EPRIME   (NEDGES + NNODES)   // edges + self loops
#define FIN      128
#define FE       16
#define HID      64
#define C1       128                 // heads=2 * 64
#define C2       64                  // heads=1 * 64
#define NG       128
#define NEG_SLOPE 0.2f

typedef __attribute__((ext_vector_type(16))) _Float16 v16h;
typedef __attribute__((ext_vector_type(8)))  _Float16 v8h;
typedef __attribute__((ext_vector_type(8)))  float    v8f;

// ordered-uint encoding so unsigned atomicMax == float max (handles negatives)
__device__ __forceinline__ unsigned f2ord(float f) {
  unsigned u = __float_as_uint(f);
  return (u & 0x80000000u) ? ~u : (u | 0x80000000u);
}
__device__ __forceinline__ float ord2f(unsigned u) {
  return (u & 0x80000000u) ? __uint_as_float(u & 0x7fffffffu)
                           : __uint_as_float(~u);
}
#define ORD_NEG_INF 0x007FFFFFu      // f2ord(-inf)

// ---------------- utility kernels ----------------
__global__ void k_fill_u32(unsigned* __restrict__ p, unsigned v, long long n) {
  long long i = blockIdx.x * (long long)blockDim.x + threadIdx.x;
  if (i < n) p[i] = v;
}

__global__ void k_f32_to_f16(const float* __restrict__ s, _Float16* __restrict__ d,
                             long long n) {
  long long i = blockIdx.x * (long long)blockDim.x + threadIdx.x;
  if (i < n) d[i] = (_Float16)s[i];
}

// convert + transpose weights: Wt[n*K + k] = (f16) W[k*Nn + n]
__global__ void k_w_transpose_f16(const float* __restrict__ W, _Float16* __restrict__ Wt,
                                  int K, int Nn) {
  long long i = blockIdx.x * (long long)blockDim.x + threadIdx.x;
  if (i >= (long long)K * Nn) return;
  int k = (int)(i / Nn), n = (int)(i % Nn);
  Wt[(size_t)n * K + k] = (_Float16)W[i];
}

// per-feature sum of edge_attr (for self-loop mean fill)
__global__ void k_edge_feat_sum(const float* __restrict__ ea, float* __restrict__ fillsum) {
  int j = threadIdx.x & 15;                       // feature
  long long lane   = blockIdx.x * (long long)(blockDim.x >> 4) + (threadIdx.x >> 4);
  long long stride = (long long)gridDim.x * (blockDim.x >> 4);
  float s = 0.f;
  for (long long e = lane; e < NEDGES; e += stride) s += ea[e * FE + j];
  atomicAdd(&fillsum[j], s);
}

// tiny single-thread precompute: fill mean, q = We.a_e, self-loop logits
__global__ void k_precompute(const float* __restrict__ fillsum,
                             const float* __restrict__ We1, const float* __restrict__ a1e,
                             const float* __restrict__ We2, const float* __restrict__ a2e,
                             float* __restrict__ fill, float* __restrict__ q1,
                             float* __restrict__ q2, float* __restrict__ alloop) {
  for (int j = 0; j < FE; ++j) fill[j] = fillsum[j] / (float)NEDGES;
  for (int j = 0; j < FE; ++j)
    for (int h = 0; h < 2; ++h) {
      float s = 0.f;
      for (int c = 0; c < HID; ++c) s += We1[j * C1 + h * HID + c] * a1e[h * HID + c];
      q1[j * 2 + h] = s;
    }
  for (int j = 0; j < FE; ++j) {
    float s = 0.f;
    for (int c = 0; c < HID; ++c) s += We2[j * C2 + c] * a2e[c];
    q2[j] = s;
  }
  float l0 = 0.f, l1 = 0.f, l2 = 0.f;
  for (int j = 0; j < FE; ++j) {
    l0 += fill[j] * q1[j * 2 + 0];
    l1 += fill[j] * q1[j * 2 + 1];
    l2 += fill[j] * q2[j];
  }
  alloop[0] = l0; alloop[1] = l1; alloop[2] = l2;
}

// ---------------- WMMA GEMM: C[M,Nn] = A[M,K](f16) x B[K,Nn](f16, given transposed) ----
// wave32; each wave computes a 16 x (NT*16) strip: A fragment reused NT times per k-step.
// 8 waves/block cover 8 consecutive M tiles. K is compile-time -> fully unrolled.
// Fragment layouts per CDNA5 ISA 7.12.2:
//   A 16x32 f16 : lane&15 = row M; lanes 0-15 hold K {0..7,16..23}, lanes 16-31 {8..15,24..31}
//   B 32x16 f16 : lane&15 = col N; lanes 0-15 hold K {0..7,16..23}, lanes 16-31 {8..15,24..31}
//   D 16x16 f32 : lane&15 = col N; VGPR r = row r (lanes 0-15) / row r+8 (lanes 16-31)
template <int K, int NT>
__global__ __launch_bounds__(256) void k_wmma_gemm(const _Float16* __restrict__ A,
                                                   const _Float16* __restrict__ Bt, // [Nn][K]
                                                   float* __restrict__ Cc,
                                                   int M, int Nn) {
  const int lane  = threadIdx.x & 31;
  const int wave  = threadIdx.x >> 5;
  const int mTile = blockIdx.x * 8 + wave;
  if (mTile * 16 >= M) return;                 // wave-uniform guard (EXEC all 1s for WMMA)
  const int rowA  = mTile * 16 + (lane & 15);
  const int nBase = blockIdx.y * (NT * 16) + (lane & 15);
  const int kh    = (lane >> 4) << 3;          // 0 or 8  (A half-row split)
  const int kb    = (lane >> 4) << 4;          // 0 or 16 (unused)
  const _Float16* arow = A + (size_t)rowA * K;

  v8f acc[NT] = {};
#pragma unroll
  for (int k0 = 0; k0 < K; k0 += 32) {
    // A fragment: two aligned 16B loads
    v8h alo = *(const v8h*)(arow + k0 + kh);
    v8h ahi = *(const v8h*)(arow + k0 + 16 + kh);
    v16h va;
#pragma unroll
    for (int j = 0; j < 8; ++j) { va[j] = alo[j]; va[8 + j] = ahi[j]; }
#pragma unroll
    for (int t = 0; t < NT; ++t) {
      // B fragment from transposed weights: K {kh..kh+7, 16+kh..16+kh+7} (two 16B loads)
      const _Float16* brow = Bt + (size_t)(nBase + t * 16) * K + k0 + kh;
      v8h blo = *(const v8h*)(brow);
      v8h bhi = *(const v8h*)(brow + 16);
      v16h vb;
#pragma unroll
      for (int j = 0; j < 8; ++j) { vb[j] = blo[j]; vb[8 + j] = bhi[j]; }
      acc[t] = __builtin_amdgcn_wmma_f32_16x16x32_f16(false, va, false, vb,
                                                      (short)0, acc[t], false, false);
    }
  }
  const int mOut = mTile * 16 + ((lane >> 4) << 3);
#pragma unroll
  for (int t = 0; t < NT; ++t) {
    const int col = blockIdx.y * (NT * 16) + t * 16 + (lane & 15);
#pragma unroll
    for (int r = 0; r < 8; ++r)
      Cc[(size_t)(mOut + r) * Nn + col] = acc[t][r];
  }
}

// ---------------- attention logits per node: al = <xw[n,h,:], a[h,:]> ----------------
__global__ void k_node_alpha(const float* __restrict__ xw,
                             const float* __restrict__ asrc, const float* __restrict__ adst,
                             float* __restrict__ alsrc, float* __restrict__ aldst, int heads) {
  long long i = blockIdx.x * (long long)blockDim.x + threadIdx.x;
  if (i >= (long long)NNODES * heads) return;
  int h = (int)(i % heads);
  long long n = i / heads;
  const float* row = xw + n * (long long)(heads * HID) + h * HID;
  float s = 0.f, d = 0.f;
  for (int c = 0; c < HID; ++c) { s += row[c] * asrc[h * HID + c]; d += row[c] * adst[h * HID + c]; }
  alsrc[i] = s; aldst[i] = d;
}

// ---------------- per-edge alpha (leaky relu) + running segment max --------------
__global__ void k_edge_alpha(const int* __restrict__ srcv, const int* __restrict__ dstv,
                             const float* __restrict__ ea,
                             const float* __restrict__ q, const float* __restrict__ alloop,
                             const float* __restrict__ alsrc, const float* __restrict__ aldst,
                             float* __restrict__ alpha, unsigned* __restrict__ amaxk,
                             int heads) {
  long long e = blockIdx.x * (long long)blockDim.x + threadIdx.x;
  if (e >= EPRIME) return;
  int s, d;
  float ale[2];
  if (e < NEDGES) {
    s = srcv[e]; d = dstv[e];
    // 16 attrs as 4 aligned float4 loads
    const float4* ev = (const float4*)(ea + e * FE);
    float4 e0 = ev[0], e1 = ev[1], e2 = ev[2], e3 = ev[3];
    float f[FE] = {e0.x,e0.y,e0.z,e0.w, e1.x,e1.y,e1.z,e1.w,
                   e2.x,e2.y,e2.z,e2.w, e3.x,e3.y,e3.z,e3.w};
    for (int h = 0; h < heads; ++h) {
      float t = 0.f;
#pragma unroll
      for (int j = 0; j < FE; ++j) t += f[j] * q[j * heads + h];
      ale[h] = t;
    }
  } else {
    s = d = (int)(e - NEDGES);
    for (int h = 0; h < heads; ++h) ale[h] = alloop[h];
  }
  for (int h = 0; h < heads; ++h) {
    float a = alsrc[(long long)s * heads + h] + aldst[(long long)d * heads + h] + ale[h];
    a = a > 0.f ? a : NEG_SLOPE * a;
    alpha[e * heads + h] = a;
    atomicMax(&amaxk[(long long)d * heads + h], f2ord(a));
  }
}

// ---------------- exp(alpha - max) + denominator accumulation ----------------
__global__ void k_edge_exp(const int* __restrict__ dstv, float* __restrict__ alpha,
                           const unsigned* __restrict__ amaxk, float* __restrict__ denom,
                           int heads) {
  long long e = blockIdx.x * (long long)blockDim.x + threadIdx.x;
  if (e >= EPRIME) return;
  int d = (e < NEDGES) ? dstv[e] : (int)(e - NEDGES);
  for (int h = 0; h < heads; ++h) {
    float m  = ord2f(amaxk[(long long)d * heads + h]);
    float ex = expf(alpha[e * heads + h] - m);
    alpha[e * heads + h] = ex;                         // reuse buffer for ex
    atomicAdd(&denom[(long long)d * heads + h], ex);
  }
}

// ---------------- weighted scatter-add aggregation (4 channels / thread) -------------
__global__ void k_aggregate4(const int* __restrict__ srcv, const int* __restrict__ dstv,
                             const float* __restrict__ xw, const float* __restrict__ ex,
                             const float* __restrict__ denom, float* __restrict__ out,
                             int heads) {
  const int C = heads * HID;
  const int Q = C >> 2;                               // float4 groups per row
  long long i = blockIdx.x * (long long)blockDim.x + threadIdx.x;
  if (i >= (long long)EPRIME * Q) return;
  long long e = i / Q;
  int qg = (int)(i % Q);
  int c0 = qg << 2;
  int h  = c0 / HID;
  int s, d;
  if (e < NEDGES) { s = srcv[e]; d = dstv[e]; } else { s = d = (int)(e - NEDGES); }
  float a = ex[e * heads + h] / (denom[(long long)d * heads + h] + 1e-16f);
  float4 v = *(const float4*)(xw + (long long)s * C + c0);
  float* o = out + (long long)d * C + c0;
  atomicAdd(o + 0, v.x * a);
  atomicAdd(o + 1, v.y * a);
  atomicAdd(o + 2, v.z * a);
  atomicAdd(o + 3, v.w * a);
}

// ---------------- bias + ELU + f16 convert (conv1 output -> conv2 GEMM input) --------
__global__ void k_bias_elu_f16(const float* __restrict__ agg, const float* __restrict__ b,
                               _Float16* __restrict__ hh, long long n, int C) {
  long long i = blockIdx.x * (long long)blockDim.x + threadIdx.x;
  if (i >= n) return;
  float v = agg[i] + b[(int)(i % C)];
  v = v > 0.f ? v : (expf(v) - 1.f);
  hh[i] = (_Float16)v;
}

// ---------------- global mean pool (bias b2 folded in, 4 channels / thread) ----------
__global__ void k_pool4(const float* __restrict__ agg2, const float* __restrict__ b2,
                        const int* __restrict__ batch, float* __restrict__ sums,
                        float* __restrict__ cnt) {
  const int Q = C2 >> 2;
  long long i = blockIdx.x * (long long)blockDim.x + threadIdx.x;
  if (i >= (long long)NNODES * Q) return;
  long long n = i / Q;
  int c0 = (int)(i % Q) << 2;
  int g = batch[n];
  float4 v = *(const float4*)(agg2 + n * C2 + c0);
  float* o = sums + g * C2 + c0;
  atomicAdd(o + 0, v.x + b2[c0 + 0]);
  atomicAdd(o + 1, v.y + b2[c0 + 1]);
  atomicAdd(o + 2, v.z + b2[c0 + 2]);
  atomicAdd(o + 3, v.w + b2[c0 + 3]);
  if (c0 == 0) atomicAdd(&cnt[g], 1.f);
}

// ---------------- MLP head: relu(emb@lw1+lb1) @ lw2 + lb2 ----------------
__global__ void k_mlp(const float* __restrict__ sums, const float* __restrict__ cnt,
                      const float* __restrict__ lw1, const float* __restrict__ lb1,
                      const float* __restrict__ lw2, const float* __restrict__ lb2,
                      float* __restrict__ out) {
  __shared__ float emb[HID];
  __shared__ float red[HID];
  int g = blockIdx.x, c = threadIdx.x;
  float ct = cnt[g]; ct = ct > 1.f ? ct : 1.f;
  emb[c] = sums[g * HID + c] / ct;
  __syncthreads();
  float z = lb1[c];
  for (int k = 0; k < HID; ++k) z += emb[k] * lw1[k * HID + c];
  z = z > 0.f ? z : 0.f;
  red[c] = z * lw2[c];
  __syncthreads();
  for (int off = HID / 2; off > 0; off >>= 1) {
    if (c < off) red[c] += red[c + off];
    __syncthreads();
  }
  if (c == 0) out[g] = red[0] + lb2[0];
}

// =======================================================================
extern "C" void kernel_launch(void* const* d_in, const int* in_sizes, int n_in,
                              void* d_out, int out_size, void* d_ws, size_t ws_size,
                              hipStream_t stream) {
  (void)in_sizes; (void)n_in; (void)out_size; (void)ws_size;
  // ---- inputs (setup_inputs order) ----
  const float* x      = (const float*)d_in[0];
  const int*   ei     = (const int*)  d_in[1];
  const float* eattr  = (const float*)d_in[2];
  const int*   batch  = (const int*)  d_in[3];
  const float* W1     = (const float*)d_in[4];
  const float* a1s    = (const float*)d_in[5];
  const float* a1d    = (const float*)d_in[6];
  const float* We1    = (const float*)d_in[7];
  const float* a1e    = (const float*)d_in[8];
  const float* b1     = (const float*)d_in[9];
  const float* W2     = (const float*)d_in[10];
  const float* a2s    = (const float*)d_in[11];
  const float* a2d    = (const float*)d_in[12];
  const float* We2    = (const float*)d_in[13];
  const float* a2e    = (const float*)d_in[14];
  const float* b2     = (const float*)d_in[15];
  const float* lw1    = (const float*)d_in[16];
  const float* lb1    = (const float*)d_in[17];
  const float* lw2    = (const float*)d_in[18];
  const float* lb2    = (const float*)d_in[19];
  float* out = (float*)d_out;
  const int* srcv = ei;
  const int* dstv = ei + NEDGES;

  // ---- workspace carve-out ----
  size_t off = 0;
  char* base = (char*)d_ws;
  auto alloc = [&](size_t bytes) -> char* {
    char* p = base + off;
    off += (bytes + 255) & ~(size_t)255;
    return p;
  };
  float*     fillsum = (float*)alloc(FE * 4);
  float*     fill    = (float*)alloc(FE * 4);
  float*     q1      = (float*)alloc(FE * 2 * 4);
  float*     q2      = (float*)alloc(FE * 4);
  float*     alloop  = (float*)alloc(4 * 4);        // [0..1]=conv1, [2]=conv2
  _Float16*  xh      = (_Float16*)alloc((size_t)NNODES * FIN * 2);
  _Float16*  w1t     = (_Float16*)alloc((size_t)C1 * FIN * 2);  // transposed [N][K]
  _Float16*  w2t     = (_Float16*)alloc((size_t)C2 * C1 * 2);   // transposed [N][K]
  float*     xw1     = (float*)alloc((size_t)NNODES * C1 * 4);
  float*     alsrc1  = (float*)alloc((size_t)NNODES * 2 * 4);
  float*     aldst1  = (float*)alloc((size_t)NNODES * 2 * 4);
  unsigned*  amaxk1  = (unsigned*)alloc((size_t)NNODES * 2 * 4);
  float*     denom1  = (float*)alloc((size_t)NNODES * 2 * 4);
  float*     alpha1  = (float*)alloc((size_t)EPRIME * 2 * 4);
  float*     agg1    = (float*)alloc((size_t)NNODES * C1 * 4);
  _Float16*  h1h     = (_Float16*)alloc((size_t)NNODES * C1 * 2);
  float*     xw2     = (float*)alloc((size_t)NNODES * C2 * 4);
  float*     alsrc2  = (float*)alloc((size_t)NNODES * 4);
  float*     aldst2  = (float*)alloc((size_t)NNODES * 4);
  unsigned*  amaxk2  = (unsigned*)alloc((size_t)NNODES * 4);
  float*     denom2  = (float*)alloc((size_t)NNODES * 4);
  float*     alpha2  = (float*)alloc((size_t)EPRIME * 4);
  float*     agg2    = (float*)alloc((size_t)NNODES * C2 * 4);
  float*     psum    = (float*)alloc((size_t)NG * C2 * 4);
  float*     pcnt    = (float*)alloc((size_t)NG * 4);

  const int T = 256;
  auto blocks = [&](long long n) { return (unsigned)((n + T - 1) / T); };

  // ---- zero / init ----
  k_fill_u32<<<blocks(FE), T, 0, stream>>>((unsigned*)fillsum, 0u, FE);
  k_fill_u32<<<blocks((long long)NNODES * 2), T, 0, stream>>>(amaxk1, ORD_NEG_INF, (long long)NNODES * 2);
  k_fill_u32<<<blocks((long long)NNODES * 2), T, 0, stream>>>((unsigned*)denom1, 0u, (long long)NNODES * 2);
  k_fill_u32<<<blocks((long long)NNODES * C1), T, 0, stream>>>((unsigned*)agg1, 0u, (long long)NNODES * C1);
  k_fill_u32<<<blocks(NNODES), T, 0, stream>>>(amaxk2, ORD_NEG_INF, NNODES);
  k_fill_u32<<<blocks(NNODES), T, 0, stream>>>((unsigned*)denom2, 0u, NNODES);
  k_fill_u32<<<blocks((long long)NNODES * C2), T, 0, stream>>>((unsigned*)agg2, 0u, (long long)NNODES * C2);
  k_fill_u32<<<blocks((long long)NG * C2), T, 0, stream>>>((unsigned*)psum, 0u, (long long)NG * C2);
  k_fill_u32<<<blocks(NG), T, 0, stream>>>((unsigned*)pcnt, 0u, NG);

  // ---- self-loop fill mean + folded edge-attention vectors ----
  k_edge_feat_sum<<<256, T, 0, stream>>>(eattr, fillsum);
  k_precompute<<<1, 1, 0, stream>>>(fillsum, We1, a1e, We2, a2e, fill, q1, q2, alloop);

  // ---- f16 staging for WMMA (weights transposed to [N][K]) ----
  k_f32_to_f16<<<blocks((long long)NNODES * FIN), T, 0, stream>>>(x, xh, (long long)NNODES * FIN);
  k_w_transpose_f16<<<blocks((long long)FIN * C1), T, 0, stream>>>(W1, w1t, FIN, C1);
  k_w_transpose_f16<<<blocks((long long)C1 * C2), T, 0, stream>>>(W2, w2t, C1, C2);

  // ================= conv1 (heads=2) =================
  {
    dim3 grid((NNODES / 16 + 7) / 8, C1 / 64);     // 782 x 2, 16x64 strip per wave
    k_wmma_gemm<FIN, 4><<<grid, 256, 0, stream>>>(xh, w1t, xw1, NNODES, C1);
  }
  k_node_alpha<<<blocks((long long)NNODES * 2), T, 0, stream>>>(xw1, a1s, a1d, alsrc1, aldst1, 2);
  k_edge_alpha<<<blocks(EPRIME), T, 0, stream>>>(srcv, dstv, eattr, q1, alloop,
                                                 alsrc1, aldst1, alpha1, amaxk1, 2);
  k_edge_exp<<<blocks(EPRIME), T, 0, stream>>>(dstv, alpha1, amaxk1, denom1, 2);
  k_aggregate4<<<blocks((long long)EPRIME * (C1 / 4)), T, 0, stream>>>(srcv, dstv, xw1, alpha1,
                                                                       denom1, agg1, 2);
  k_bias_elu_f16<<<blocks((long long)NNODES * C1), T, 0, stream>>>(agg1, b1, h1h,
                                                                   (long long)NNODES * C1, C1);

  // ================= conv2 (heads=1) =================
  {
    dim3 grid((NNODES / 16 + 7) / 8, C2 / 64);     // 782 x 1
    k_wmma_gemm<C1, 4><<<grid, 256, 0, stream>>>(h1h, w2t, xw2, NNODES, C2);
  }
  k_node_alpha<<<blocks(NNODES), T, 0, stream>>>(xw2, a2s, a2d, alsrc2, aldst2, 1);
  k_edge_alpha<<<blocks(EPRIME), T, 0, stream>>>(srcv, dstv, eattr, q2, alloop + 2,
                                                 alsrc2, aldst2, alpha2, amaxk2, 1);
  k_edge_exp<<<blocks(EPRIME), T, 0, stream>>>(dstv, alpha2, amaxk2, denom2, 1);
  k_aggregate4<<<blocks((long long)EPRIME * (C2 / 4)), T, 0, stream>>>(srcv, dstv, xw2, alpha2,
                                                                       denom2, agg2, 1);

  // ================= pool + MLP head =================
  k_pool4<<<blocks((long long)NNODES * (C2 / 4)), T, 0, stream>>>(agg2, b2, batch, psum, pcnt);
  k_mlp<<<NG, HID, 0, stream>>>(psum, pcnt, lw1, lb1, lw2, lb2, out);
}
